// ProteinStudentModel_57870389347021
// MI455X (gfx1250) — compile-verified, matrix operand order checked by reading, not run on hardware
//
#include <hip/hip_runtime.h>
#include <math.h>

// ---------------- problem constants ----------------
#define NN    30000
#define EE    960000
#define GG    64
#define DIN   2560
#define HIDN  256

typedef __attribute__((ext_vector_type(16))) __bf16 v16bf;
typedef __attribute__((ext_vector_type(2)))  __bf16 v2bf;
typedef __attribute__((ext_vector_type(8)))  float  v8f;

// pack two fp32 -> packed 2x bf16 dword (low = a): single v_cvt_pk_bf16_f32
static __device__ __forceinline__ unsigned pk2bf(float a, float b) {
#if defined(__has_builtin) && __has_builtin(__builtin_amdgcn_cvt_pk_bf16_f32)
  v2bf r = __builtin_amdgcn_cvt_pk_bf16_f32(a, b);
  return __builtin_bit_cast(unsigned, r);
#else
  v2bf r;
  r[0] = (__bf16)a;
  r[1] = (__bf16)b;
  return __builtin_bit_cast(unsigned, r);
#endif
}

static __device__ __forceinline__ float wred_sum(float v) {
  #pragma unroll
  for (int o = 16; o; o >>= 1) v += __shfl_xor(v, o, 32);
  return v;
}
static __device__ __forceinline__ float wred_max(float v) {
  #pragma unroll
  for (int o = 16; o; o >>= 1) v = fmaxf(v, __shfl_xor(v, o, 32));
  return v;
}

// ---------------- tiny utility kernels ----------------
__global__ void zero_f_kernel(float* p, int n) {
  for (int i = blockIdx.x * blockDim.x + threadIdx.x; i < n; i += gridDim.x * blockDim.x)
    p[i] = 0.0f;
}
__global__ void zero_i_kernel(int* p, int n) {
  for (int i = blockIdx.x * blockDim.x + threadIdx.x; i < n; i += gridDim.x * blockDim.x)
    p[i] = 0;
}

// vfeat[j] = virtual @ W_vp + b_vp   (1x2574 @ 2574x14)
__global__ void vfeat_kernel(const float* __restrict__ virt, const float* __restrict__ Wvp,
                             const float* __restrict__ bvp, float* __restrict__ vfeat) {
  int j = threadIdx.x;
  if (j < 14) {
    float s = bvp[j];
    for (int i = 0; i < 2574; ++i) s += virt[i] * Wvp[i * 14 + j];
    vfeat[j] = s;
  }
}

// Mmat[i][j] = sum_k W_dp[i,k] * W_eh[k,j]   (14x2560 @ 2560x256)
__global__ void mmat_kernel(const float* __restrict__ Wdp, const float* __restrict__ Weh,
                            float* __restrict__ Mmat) {
  int i = blockIdx.x, j = threadIdx.x;
  float s = 0.0f;
  for (int k = 0; k < DIN; ++k) s += Wdp[(size_t)i * DIN + k] * Weh[(size_t)k * HIDN + j];
  Mmat[i * HIDN + j] = s;
}

// cvec[j] = vfeat @ Mmat + b_dp @ W_eh + b_eh
__global__ void cvec_kernel(const float* __restrict__ vfeat, const float* __restrict__ Mmat,
                            const float* __restrict__ bdp, const float* __restrict__ Weh,
                            const float* __restrict__ beh, float* __restrict__ cvec) {
  int j = threadIdx.x;
  float s = beh[j];
  for (int i = 0; i < 14; ++i) s += vfeat[i] * Mmat[i * HIDN + j];
  for (int k = 0; k < DIN; ++k) s += bdp[k] * Weh[(size_t)k * HIDN + j];
  cvec[j] = s;
}

// ---------------- CSR build (dst-sorted edge buckets, self loops appended) ----------------
__global__ void count_kernel(const int* __restrict__ ei, int* __restrict__ cnt) {
  int total = EE + NN;
  for (int i = blockIdx.x * blockDim.x + threadIdx.x; i < total; i += gridDim.x * blockDim.x) {
    int d = (i < EE) ? ei[EE + i] : (i - EE);
    atomicAdd(&cnt[d], 1);
  }
}

__global__ __launch_bounds__(1024)
void scan_kernel(const int* __restrict__ cnt, int* __restrict__ offs, int n) {
  __shared__ int tmp[1024];
  __shared__ int carry;
  int tid = threadIdx.x;
  if (tid == 0) carry = 0;
  __syncthreads();
  for (int base = 0; base < n; base += 1024) {
    int i = base + tid;
    int v = (i < n) ? cnt[i] : 0;
    tmp[tid] = v;
    __syncthreads();
    for (int off = 1; off < 1024; off <<= 1) {
      int t = (tid >= off) ? tmp[tid - off] : 0;
      __syncthreads();
      tmp[tid] += t;
      __syncthreads();
    }
    if (i < n) offs[i] = carry + tmp[tid] - v;   // exclusive
    __syncthreads();
    if (tid == 0) carry += tmp[1023];
    __syncthreads();
  }
  if (tid == 0) offs[n] = carry;
}

__global__ void copy_i_kernel(const int* __restrict__ a, int* __restrict__ b, int n) {
  for (int i = blockIdx.x * blockDim.x + threadIdx.x; i < n; i += gridDim.x * blockDim.x)
    b[i] = a[i];
}

__global__ void scatter_kernel(const int* __restrict__ ei, int* __restrict__ cursor,
                               int* __restrict__ srcs) {
  int total = EE + NN;
  for (int i = blockIdx.x * blockDim.x + threadIdx.x; i < total; i += gridDim.x * blockDim.x) {
    int d, s;
    if (i < EE) { s = ei[i]; d = ei[EE + i]; } else { s = i - EE; d = i - EE; }
    int p = atomicAdd(&cursor[d], 1);
    srcs[p] = s;
  }
}

// ---------------- WMMA GEMM: D = act(A[M,K] @ B[K,Nc] (+ fused extras)) ----------------
// mode 0: D = relu(A@B + dssp@Mmat + cvec)   (input projection)
// mode 1: D = A@B                            (GAT xp projection)
#define TM 128
#define TN 64
#define TK 64
#define LP 8   // LDS row padding (bf16): pitch 72 elems = 144B (16B aligned, bank-spread)

__global__ __launch_bounds__(256)
void gemm_wmma_kernel(const float* __restrict__ A, const float* __restrict__ B,
                      float* __restrict__ D, int Mrows, int Kdim, int Ncols,
                      int mode, const float* __restrict__ dssp,
                      const float* __restrict__ Mmat, const float* __restrict__ cvec) {
  // A tile row-major; B tile TRANSPOSED (BsT[col][k]) so per-lane fragments are
  // 32 contiguous bytes -> ds_load_b128 pairs, no scalar packing.
  __shared__ __bf16 As[TM][TK + LP];
  __shared__ __bf16 BsT[TN][TK + LP];

  const int tid  = threadIdx.x;
  const int lane = tid & 31;
  const int w    = tid >> 5;      // 8 waves
  const int wm   = w & 3;         // 4 row groups of 32
  const int wn   = w >> 2;        // 2 col groups of 32
  const int m0   = blockIdx.x * TM;
  const int n0   = blockIdx.y * TN;
  const int hl   = lane >> 4;     // half (0/1)
  const int ll   = lane & 15;
  const bool fullM = (m0 + TM <= Mrows);

  v8f acc[2][2];
  #pragma unroll
  for (int i = 0; i < 2; ++i)
    #pragma unroll
    for (int j = 0; j < 2; ++j)
      #pragma unroll
      for (int e = 0; e < 8; ++e) acc[i][j][e] = 0.0f;

  const int kTiles = Kdim / TK;
  for (int kt = 0; kt < kTiles; ++kt) {
    // ---- stage A tile 128x64 (fp32 -> packed bf16 dword stores) ----
    if (fullM) {
      #pragma unroll
      for (int it = 0; it < 8; ++it) {
        int i = it * 256 + tid;            // 2048 float4 slots
        int r = i >> 4;                    // 16 float4 per 64-col row
        int c = (i & 15) * 4;
        const float* ap = A + (size_t)(m0 + r) * Kdim + (size_t)kt * TK + c;
        float4 v = *(const float4*)ap;
        __builtin_prefetch(ap + TK, 0, 1);   // speculative; OOB dropped by HW
        unsigned* dst = (unsigned*)&As[r][c];
        dst[0] = pk2bf(v.x, v.y);
        dst[1] = pk2bf(v.z, v.w);
      }
    } else {
      #pragma unroll
      for (int it = 0; it < 8; ++it) {
        int i = it * 256 + tid;
        int r = i >> 4;
        int c = (i & 15) * 4;
        int gr = m0 + r;
        float4 v = make_float4(0.f, 0.f, 0.f, 0.f);
        if (gr < Mrows) v = *(const float4*)(A + (size_t)gr * Kdim + (size_t)kt * TK + c);
        unsigned* dst = (unsigned*)&As[r][c];
        dst[0] = pk2bf(v.x, v.y);
        dst[1] = pk2bf(v.z, v.w);
      }
    }
    // ---- stage B tile 64x64, transposed into BsT[col][k] ----
    #pragma unroll
    for (int it = 0; it < 8; ++it) {
      int i = it * 256 + tid;              // 2048 k-pairs
      int col = i & 63;
      int k   = (i >> 6) * 2;
      const float* bp = B + (size_t)(kt * TK + k) * Ncols + n0 + col;
      float b0 = bp[0];
      float b1 = bp[Ncols];
      *(unsigned*)&BsT[col][k] = pk2bf(b0, b1);
    }
    __syncthreads();

    // ---- 2 k-steps of 32: fragments are 32 contiguous bytes (2x ds_load_b128) ----
    #pragma unroll
    for (int ks = 0; ks < 2; ++ks) {
      v16bf af[2], bfv[2];
      #pragma unroll
      for (int mi = 0; mi < 2; ++mi) {
        int row = wm * 32 + mi * 16 + ll;
        __builtin_memcpy(&af[mi], &As[row][ks * 32 + hl * 16], 32);
      }
      #pragma unroll
      for (int ni = 0; ni < 2; ++ni) {
        int col = wn * 32 + ni * 16 + ll;
        __builtin_memcpy(&bfv[ni], &BsT[col][ks * 32 + hl * 16], 32);
      }
      #pragma unroll
      for (int mi = 0; mi < 2; ++mi)
        #pragma unroll
        for (int ni = 0; ni < 2; ++ni)
          acc[mi][ni] = __builtin_amdgcn_wmma_f32_16x16x32_bf16(
              false, af[mi], false, bfv[ni], (short)0, acc[mi][ni], false, false);
    }
    __syncthreads();
  }

  // ---- epilogue: C/D layout lane->col=ll, row = hl*8 + r ----
  #pragma unroll
  for (int mi = 0; mi < 2; ++mi) {
    #pragma unroll
    for (int ni = 0; ni < 2; ++ni) {
      int coln = n0 + wn * 32 + ni * 16 + ll;
      float mmcol[14];
      float cadd = 0.0f;
      if (mode == 0) {
        #pragma unroll
        for (int i = 0; i < 14; ++i) mmcol[i] = Mmat[i * HIDN + coln];
        cadd = cvec[coln];
      }
      #pragma unroll
      for (int r = 0; r < 8; ++r) {
        int rowm = m0 + wm * 32 + mi * 16 + hl * 8 + r;
        if (rowm >= Mrows) continue;
        float v = acc[mi][ni][r];
        if (mode == 0) {
          const float* drow = dssp + (size_t)rowm * 14;
          float extra = 0.0f;
          #pragma unroll
          for (int i = 0; i < 14; ++i) extra += drow[i] * mmcol[i];
          v = fmaxf(v + extra + cadd, 0.0f);
        }
        D[(size_t)rowm * Ncols + coln] = v;
      }
    }
  }
}

// ---------------- attention logits: a_s/a_d [N,H] from xp [N,H*256] ----------------
template<int H>
__global__ void att_kernel(const float* __restrict__ xp, const float* __restrict__ as_w,
                           const float* __restrict__ ad_w, float* __restrict__ a_s,
                           float* __restrict__ a_d) {
  int h = threadIdx.x >> 5;
  int lane = threadIdx.x & 31;
  int n = blockIdx.x;
  const float* row = xp + (size_t)n * (H * 256) + h * 256;
  float s1 = 0.f, s2 = 0.f;
  for (int c = lane; c < 256; c += 32) {
    float v = row[c];
    s1 += v * as_w[h * 256 + c];
    s2 += v * ad_w[h * 256 + c];
  }
  s1 = wred_sum(s1);
  s2 = wred_sum(s2);
  if (lane == 0) { a_s[n * H + h] = s1; a_d[n * H + h] = s2; }
}

// ---------------- fused GAT aggregation + head-mean + bias + residual + LN + relu -------
// one wave per destination node; lanes carry the H*256 accumulator (feature f = k*32+lane)
template<int H>
__global__ __launch_bounds__(256)
void agg_kernel(const float* __restrict__ xp, const int* __restrict__ srcs,
                const int* __restrict__ offs, const float* __restrict__ a_s,
                const float* __restrict__ a_d, const float* __restrict__ gbias,
                const float* __restrict__ ng, const float* __restrict__ nb,
                float* __restrict__ x, int n_nodes) {
  const int lane = threadIdx.x & 31;
  const int wid  = threadIdx.x >> 5;
  const int n    = blockIdx.x * 8 + wid;
  if (n >= n_nodes) return;

  const int HC = H * 256;
  const int RK = H * 8;
  const int beg = offs[n], end = offs[n + 1];

  float adn[H];
  #pragma unroll
  for (int h = 0; h < H; ++h) adn[h] = a_d[n * H + h];

  // pass 1: per-head max of leaky(alpha), edge-parallel across lanes
  float mx[H];
  #pragma unroll
  for (int h = 0; h < H; ++h) mx[h] = -3.4e38f;
  for (int e = beg + lane; e < end; e += 32) {
    int s = srcs[e];
    #pragma unroll
    for (int h = 0; h < H; ++h) {
      float v = a_s[s * H + h] + adn[h];
      v = (v > 0.f) ? v : 0.2f * v;
      mx[h] = fmaxf(mx[h], v);
    }
  }
  #pragma unroll
  for (int h = 0; h < H; ++h) mx[h] = wred_max(mx[h]);

  // pass 2: softmax weights + weighted gather of xp rows (feature-parallel)
  float ssum[H];
  #pragma unroll
  for (int h = 0; h < H; ++h) ssum[h] = 0.f;
  float acc[RK];
  #pragma unroll
  for (int k = 0; k < RK; ++k) acc[k] = 0.f;

  for (int e = beg; e < end; ++e) {
    int s = srcs[e];
    float ea[H];
    #pragma unroll
    for (int h = 0; h < H; ++h) {
      float v = a_s[s * H + h] + adn[h];
      v = (v > 0.f) ? v : 0.2f * v;
      ea[h] = __expf(v - mx[h]);
      ssum[h] += ea[h];
    }
    const float* xr = xp + (size_t)s * HC;
    #pragma unroll
    for (int k = 0; k < RK; ++k) acc[k] += ea[k >> 3] * xr[k * 32 + lane];
  }

  float inv[H];
  #pragma unroll
  for (int h = 0; h < H; ++h) inv[h] = 1.0f / ssum[h];

  // head-mean + bias + residual
  float y[8];
  float s1 = 0.f, s2 = 0.f;
  #pragma unroll
  for (int k0 = 0; k0 < 8; ++k0) {
    int c = k0 * 32 + lane;
    float v = 0.f;
    #pragma unroll
    for (int h = 0; h < H; ++h) v += acc[k0 + 8 * h] * inv[h];
    v = v * (1.0f / H) + gbias[c] + x[(size_t)n * 256 + c];
    y[k0] = v;
    s1 += v;
    s2 += v * v;
  }
  s1 = wred_sum(s1);
  s2 = wred_sum(s2);
  float mean = s1 * (1.0f / 256.0f);
  float var  = fmaxf(s2 * (1.0f / 256.0f) - mean * mean, 0.0f);
  float rstd = rsqrtf(var + 1e-5f);
  #pragma unroll
  for (int k0 = 0; k0 < 8; ++k0) {
    int c = k0 * 32 + lane;
    float v = (y[k0] - mean) * rstd * ng[c] + nb[c];
    x[(size_t)n * 256 + c] = fmaxf(v, 0.0f);
  }
}

// ---------------- pooling + MLP ----------------
__global__ void pool_kernel(const float* __restrict__ x, const int* __restrict__ batch,
                            float* __restrict__ gsum, float* __restrict__ gcount) {
  int n = blockIdx.x, c = threadIdx.x;
  int b = batch[n];
  atomicAdd(&gsum[b * 256 + c], x[(size_t)n * 256 + c]);
  if (c == 0) atomicAdd(&gcount[b], 1.0f);
}

__global__ void mlp1_kernel(const float* __restrict__ gsum, const float* __restrict__ gcount,
                            const float* __restrict__ W1, const float* __restrict__ b1,
                            float* __restrict__ h1) {
  int g = blockIdx.x, j = threadIdx.x;
  float invc = 1.0f / fmaxf(gcount[g], 1.0f);
  float s = b1[j];
  for (int k = 0; k < 256; ++k) s += gsum[g * 256 + k] * invc * W1[k * 256 + j];
  h1[g * 256 + j] = fmaxf(s, 0.0f);
}

__global__ void mlp2_kernel(const float* __restrict__ h1, const float* __restrict__ W2,
                            const float* __restrict__ b2, float* __restrict__ out) {
  int g = blockIdx.x, j = threadIdx.x;
  float s = b2[j];
  for (int k = 0; k < 256; ++k) s += h1[g * 256 + k] * W2[k * 256 + j];
  out[g * 256 + j] = s;
}

// ---------------- host launch ----------------
extern "C" void kernel_launch(void* const* d_in, const int* in_sizes, int n_in,
                              void* d_out, int out_size, void* d_ws, size_t ws_size,
                              hipStream_t stream) {
  const float* esm   = (const float*)d_in[0];
  const float* dssp  = (const float*)d_in[1];
  const float* virt  = (const float*)d_in[2];
  const int*   ei    = (const int*)  d_in[3];
  const int*   batch = (const int*)  d_in[4];
  const float* W_vp  = (const float*)d_in[5];
  const float* b_vp  = (const float*)d_in[6];
  const float* W_dp  = (const float*)d_in[7];
  const float* b_dp  = (const float*)d_in[8];
  const float* W_eh  = (const float*)d_in[9];
  const float* b_eh  = (const float*)d_in[10];

  struct GatP { const float *W, *as, *ad, *b, *ng, *nb; int H; };
  GatP L[3] = {
    {(const float*)d_in[11], (const float*)d_in[12], (const float*)d_in[13],
     (const float*)d_in[14], (const float*)d_in[15], (const float*)d_in[16], 4},
    {(const float*)d_in[17], (const float*)d_in[18], (const float*)d_in[19],
     (const float*)d_in[20], (const float*)d_in[21], (const float*)d_in[22], 4},
    {(const float*)d_in[23], (const float*)d_in[24], (const float*)d_in[25],
     (const float*)d_in[26], (const float*)d_in[27], (const float*)d_in[28], 1},
  };
  const float* mlp_W1 = (const float*)d_in[29];
  const float* mlp_b1 = (const float*)d_in[30];
  const float* mlp_W2 = (const float*)d_in[31];
  const float* mlp_b2 = (const float*)d_in[32];

  float* out_enh = (float*)d_out;             // [64,256]
  float* xbuf    = (float*)d_out + GG * HIDN; // [N,256] second output, used in-place

  // ---- workspace layout (floats then ints) ----
  float* wsf = (float*)d_ws;
  float* vfeat  = wsf;                     // 16
  float* Mmat   = wsf + 16;                // 14*256 = 3584
  float* cvec   = wsf + 3712;              // 256
  float* a_s    = wsf + 4096;              // N*4
  float* a_d    = wsf + 4096 + NN * 4;     // N*4
  float* gsum   = wsf + 4096 + NN * 8;     // 64*256
  float* gcount = gsum + GG * HIDN;        // 64
  float* h1     = gcount + 64;             // 64*256
  float* xp     = wsf + 4096 + NN * 8 + 2 * GG * HIDN + 256; // N*1024
  int*   wsi    = (int*)(xp + (size_t)NN * 1024);
  int*   cnts   = wsi;                     // N
  int*   offs   = wsi + NN + 64;           // N+1
  int*   cursor = offs + NN + 64;          // N
  int*   srcs   = cursor + NN + 64;        // E+N

  // ---- build CSR (every call; workspace is not preserved across replays) ----
  zero_i_kernel<<<256, 256, 0, stream>>>(cnts, NN);
  count_kernel<<<2048, 256, 0, stream>>>(ei, cnts);
  scan_kernel<<<1, 1024, 0, stream>>>(cnts, offs, NN);
  copy_i_kernel<<<256, 256, 0, stream>>>(offs, cursor, NN);
  scatter_kernel<<<2048, 256, 0, stream>>>(ei, cursor, srcs);

  // ---- collapsed input projection constants ----
  vfeat_kernel<<<1, 16, 0, stream>>>(virt, W_vp, b_vp, vfeat);
  mmat_kernel<<<14, 256, 0, stream>>>(W_dp, W_eh, Mmat);
  cvec_kernel<<<1, 256, 0, stream>>>(vfeat, Mmat, b_dp, W_eh, b_eh, cvec);

  // ---- h0 = relu(esm@W_eh + dssp@Mmat + cvec) : one big WMMA GEMM ----
  {
    dim3 grid((NN + TM - 1) / TM, HIDN / TN);
    gemm_wmma_kernel<<<grid, 256, 0, stream>>>(esm, W_eh, xbuf, NN, DIN, HIDN,
                                               0, dssp, Mmat, cvec);
  }

  // ---- 3 GAT layers ----
  for (int l = 0; l < 3; ++l) {
    int H = L[l].H;
    int HC = H * HIDN;
    dim3 grid((NN + TM - 1) / TM, HC / TN);
    gemm_wmma_kernel<<<grid, 256, 0, stream>>>(xbuf, L[l].W, xp, NN, HIDN, HC,
                                               1, nullptr, nullptr, nullptr);
    if (H == 4) {
      att_kernel<4><<<NN, 128, 0, stream>>>(xp, L[l].as, L[l].ad, a_s, a_d);
      agg_kernel<4><<<(NN + 7) / 8, 256, 0, stream>>>(xp, srcs, offs, a_s, a_d,
                                                      L[l].b, L[l].ng, L[l].nb, xbuf, NN);
    } else {
      att_kernel<1><<<NN, 32, 0, stream>>>(xp, L[l].as, L[l].ad, a_s, a_d);
      agg_kernel<1><<<(NN + 7) / 8, 256, 0, stream>>>(xp, srcs, offs, a_s, a_d,
                                                      L[l].b, L[l].ng, L[l].nb, xbuf, NN);
    }
  }

  // ---- mean pooling + MLP ----
  zero_f_kernel<<<64, 256, 0, stream>>>(gsum, GG * HIDN + GG);
  pool_kernel<<<NN, 256, 0, stream>>>(xbuf, batch, gsum, gcount);
  mlp1_kernel<<<GG, 256, 0, stream>>>(gsum, gcount, mlp_W1, mlp_b1, h1);
  mlp2_kernel<<<GG, 256, 0, stream>>>(h1, mlp_W2, mlp_b2, out_enh);
}
